// EfficientDsConv_47090021433368
// MI455X (gfx1250) — matrix operand order repaired and checked by reading, not compiled, not run on hardware
//
#include <hip/hip_runtime.h>

// ---------------------------------------------------------------------------
// EfficientDsConv on gfx1250: LN -> per-head causal Toeplitz matmul via WMMA
// out[b,i,h,c] = residual + sum_{s<=i} u[b,s,h,c] * krev_h[i-s]
// ---------------------------------------------------------------------------

typedef __attribute__((ext_vector_type(16))) __bf16 v16bf;
typedef __attribute__((ext_vector_type(8)))  float  v8f;

struct U4 { unsigned int a, b, c, d; };          // trivial 16B chunk (union-safe)

union FragBF {
    unsigned short u[16];
    U4             q[2];
    v16bf          v;
};

#if __has_builtin(__builtin_amdgcn_global_load_async_to_lds_b128) && \
    __has_builtin(__builtin_amdgcn_s_wait_asynccnt)
#define USE_ASYNC 1
typedef __attribute__((ext_vector_type(4))) int v4i;
typedef __attribute__((address_space(1))) v4i* gas_t;   // global int4*
typedef __attribute__((address_space(3))) v4i* las_t;   // LDS int4*
#define ASYNC_CP_B128(gsrc, ldst)                                             \
    __builtin_amdgcn_global_load_async_to_lds_b128((gas_t)(gsrc), (las_t)(ldst), 0, 0)
#else
#define USE_ASYNC 0
#endif

#define SEQ   8192
#define DIMS  1024
#define NHEAD 16
#define NCOLS 128      // 2 batches * 64 channels

__device__ __forceinline__ unsigned short f2bf(float f) {
    unsigned int u = __float_as_uint(f);
    unsigned int r = u + 0x7FFFu + ((u >> 16) & 1u);   // round-to-nearest-even
    return (unsigned short)(r >> 16);
}

// ---------------------------------------------------------------------------
// Kernel 1: LayerNorm. One block per (b,s) row of 1024.
// Writes: d_out = residual (f32), u_bf = bf16 u in [head][s][b*64+c] layout.
// ---------------------------------------------------------------------------
__global__ __launch_bounds__(256)
void ln_kernel(const float* __restrict__ x, const float* __restrict__ gamma,
               const float* __restrict__ beta, const float* __restrict__ pD,
               float* __restrict__ out, unsigned short* __restrict__ u_bf) {
    const int row = blockIdx.x;                 // b*SEQ + s
    const int t   = threadIdx.x;                // 256
    const int d   = t * 4;

    const float4 v = ((const float4*)(x + (size_t)row * DIMS))[t];
    float s = v.x + v.y + v.z + v.w;
    float q = v.x * v.x + v.y * v.y + v.z * v.z + v.w * v.w;
#pragma unroll
    for (int off = 16; off > 0; off >>= 1) {
        s += __shfl_down(s, off, 32);
        q += __shfl_down(q, off, 32);
    }
    __shared__ float rs[8], rq[8], mv[2];
    if ((t & 31) == 0) { rs[t >> 5] = s; rq[t >> 5] = q; }
    __syncthreads();
    if (t == 0) {
        float S = 0.f, Q = 0.f;
#pragma unroll
        for (int i = 0; i < 8; ++i) { S += rs[i]; Q += rq[i]; }
        float m   = S * (1.0f / DIMS);
        float var = Q * (1.0f / DIMS) - m * m;
        mv[0] = m;
        mv[1] = rsqrtf(var + 1e-5f);
    }
    __syncthreads();
    const float m = mv[0], r = mv[1];

    float u0 = (v.x - m) * r * gamma[d + 0] + beta[d + 0];
    float u1 = (v.y - m) * r * gamma[d + 1] + beta[d + 1];
    float u2 = (v.z - m) * r * gamma[d + 2] + beta[d + 2];
    float u3 = (v.w - m) * r * gamma[d + 3] + beta[d + 3];

    float4 res;
    res.x = u0 * pD[d + 0]; res.y = u1 * pD[d + 1];
    res.z = u2 * pD[d + 2]; res.w = u3 * pD[d + 3];
    ((float4*)(out + (size_t)row * DIMS))[t] = res;

    const int b = row >> 13, srow = row & (SEQ - 1);
    const int h = d >> 6, c = d & 63;
    const size_t idx = ((size_t)(h * SEQ + srow)) * NCOLS + b * 64 + c;  // mult of 4
    unsigned int lo = (unsigned int)f2bf(u0) | ((unsigned int)f2bf(u1) << 16);
    unsigned int hi = (unsigned int)f2bf(u2) | ((unsigned int)f2bf(u3) << 16);
    ((unsigned int*)u_bf)[idx / 2 + 0] = lo;
    ((unsigned int*)u_bf)[idx / 2 + 1] = hi;
}

// ---------------------------------------------------------------------------
// Kernel 2: reverse + bf16-convert the conv kernel: krev[h][j] = W[8191-j][h]
// ---------------------------------------------------------------------------
__global__ __launch_bounds__(256)
void krev_kernel(const float* __restrict__ w, unsigned short* __restrict__ krev) {
    const int f = blockIdx.x * 256 + threadIdx.x;   // < SEQ*NHEAD
    const int h = f & 15, j = f >> 4;
    krev[h * SEQ + j] = f2bf(w[(size_t)(SEQ - 1 - j) * NHEAD + h]);
}

// ---------------------------------------------------------------------------
// Kernel 3: per-head triangular Toeplitz GEMM via WMMA bf16.
// grid = (64 row-tiles, 16 heads), block = 256 (8 waves).
// Wave w owns rows [i0+16w, i0+16w+16) x all 128 cols (8 WMMA n-tiles).
// ---------------------------------------------------------------------------
__global__ __launch_bounds__(256)
void conv_kernel(const unsigned short* __restrict__ u_bf,
                 const unsigned short* __restrict__ krev,
                 float* __restrict__ out) {
    __shared__ __align__(16) unsigned short s_krev[SEQ];          // 16 KB
    __shared__ __align__(16) unsigned short s_b[2][32 * NCOLS];   // 2 x 8 KB

    const int h   = blockIdx.y;
    const int rt  = blockIdx.x;
    const int i0  = rt * 128;
    const int t   = threadIdx.x;
    const int lane = t & 31, w = t >> 5;
    const int grp  = lane >> 4;
    const int nkt  = rt * 4 + 4;                 // triangular: s0 <= i0+127

    // ---- stage the whole reversed kernel for this head (32 u16 per thread)
    {
        const unsigned short* g = krev + h * SEQ + t * 32;
        unsigned short* l = &s_krev[t * 32];
#if USE_ASYNC
#pragma unroll
        for (int qi = 0; qi < 4; ++qi)
            ASYNC_CP_B128(g + qi * 8, l + qi * 8);
#else
#pragma unroll
        for (int qi = 0; qi < 4; ++qi) ((U4*)l)[qi] = ((const U4*)g)[qi];
#endif
    }

    // ---- B-tile staging: 32 rows x 128 cols bf16 (8 KB), 32B per thread
    auto stage_b = [&](int bufn, int kt) {
        const int s0   = kt * 32;
        const int srow = t >> 3;
        const int coff = (t & 7) * 16;          // u16 elements
        const unsigned short* g =
            u_bf + ((size_t)(h * SEQ + s0 + srow)) * NCOLS + coff;
        unsigned short* l = &s_b[bufn][srow * NCOLS + coff];
#if USE_ASYNC
        ASYNC_CP_B128(g, l);
        ASYNC_CP_B128(g + 8, l + 8);
#else
        ((U4*)l)[0] = ((const U4*)g)[0];
        ((U4*)l)[1] = ((const U4*)g)[1];
#endif
    };

    stage_b(0, 0);
#if USE_ASYNC
    __builtin_amdgcn_s_wait_asynccnt(0);
#endif
    __syncthreads();

    v8f acc[8] = {};
    const int irow = i0 + w * 16 + (lane & 15);   // A-fragment row for this lane

    int buf = 0;
    for (int kt = 0; kt < nkt; ++kt) {
        if (kt + 1 < nkt) stage_b(buf ^ 1, kt + 1);
        if (kt + 2 < nkt)
            __builtin_prefetch(u_bf + ((size_t)(h * SEQ + (kt + 2) * 32 + (t >> 3)))
                                          * NCOLS + (t & 7) * 16, 0, 1);

        // ---- build 16x32 Toeplitz A fragment from LDS krev (causal guard)
        const int s0 = kt * 32;
        FragBF a;
#pragma unroll
        for (int vr = 0; vr < 8; ++vr) {
            const int kb = ((vr & 4) ? 16 : 0) + grp * 8 + (vr & 3) * 2;
#pragma unroll
            for (int e = 0; e < 2; ++e) {
                const int dd = irow - (s0 + kb + e);
                a.u[vr * 2 + e] = (dd >= 0) ? s_krev[dd] : (unsigned short)0;
            }
        }

        // ---- load ALL 8 B fragments first (distinct regs -> one dscnt wait),
        //      then issue the 8 WMMAs back-to-back so they pipeline densely.
        FragBF bfr[8];
        const unsigned short* pbase = &s_b[buf][lane & 15];
#pragma unroll
        for (int j = 0; j < 8; ++j) {
#pragma unroll
            for (int s = 0; s < 16; ++s) {
                const int kk = ((s & 8) ? 16 : 0) + grp * 8 + (s & 7);
                bfr[j].u[s] = pbase[kk * NCOLS + j * 16];
            }
        }
#pragma unroll
        for (int j = 0; j < 8; ++j) {
            acc[j] = __builtin_amdgcn_wmma_f32_16x16x32_bf16(
                false, a.v, false, bfr[j].v, (short)0, acc[j], false, false);
        }

#if USE_ASYNC
        __builtin_amdgcn_s_wait_asynccnt(0);
#endif
        __syncthreads();
        buf ^= 1;
    }

    // ---- epilogue: C/D layout -> out[b, i, h*64+c] (residual already there)
#pragma unroll
    for (int r = 0; r < 8; ++r) {
        const int orow = i0 + w * 16 + r + grp * 8;
#pragma unroll
        for (int j = 0; j < 8; ++j) {
            const int n  = j * 16 + (lane & 15);
            const int bb = n >> 6, c = n & 63;
            const size_t oidx =
                ((size_t)(bb * SEQ + orow)) * DIMS + h * 64 + c;
            out[oidx] += acc[j][r];
        }
    }
}

// ---------------------------------------------------------------------------
extern "C" void kernel_launch(void* const* d_in, const int* in_sizes, int n_in,
                              void* d_out, int out_size, void* d_ws, size_t ws_size,
                              hipStream_t stream) {
    const float* x     = (const float*)d_in[0];
    const float* gamma = (const float*)d_in[1];
    const float* beta  = (const float*)d_in[2];
    const float* wgt   = (const float*)d_in[3];
    const float* pD    = (const float*)d_in[4];
    float* out = (float*)d_out;

    unsigned short* u_bf = (unsigned short*)d_ws;                       // 32 MB
    unsigned short* krev =
        (unsigned short*)((char*)d_ws + (size_t)NHEAD * SEQ * NCOLS * 2);  // 256 KB

    ln_kernel<<<2 * SEQ, 256, 0, stream>>>(x, gamma, beta, pD, out, u_bf);
    krev_kernel<<<(SEQ * NHEAD) / 256, 256, 0, stream>>>(wgt, krev);

    dim3 grid(64, NHEAD);
    conv_kernel<<<grid, 256, 0, stream>>>(u_bf, krev, out);
}
